// CLARA_Enhanced_59536836657886
// MI455X (gfx1250) — matrix operand order repaired and checked by reading, not compiled, not run on hardware
//
#include <hip/hip_runtime.h>

namespace {

constexpr int ED = 256, FD = 512, RD = 128;
constexpr int MT = 16;          // rows per block (per modality)
constexpr int NWAVES = 8;       // 256 threads, wave32
constexpr int LDS_S = 260;      // f32 residual-stream stride (260 % 64 = 4 banks)
constexpr int LDT   = 516;      // f32 scratch stride
constexpr int LDA   = 264;      // bf16 activation stride (528B = 33*16B, 16B-aligned rows)
constexpr int LDH   = 520;      // bf16 hidden stride (1040B = 65*16B)

typedef __attribute__((ext_vector_type(16))) __bf16 v16bf;
typedef __attribute__((ext_vector_type(8)))  __bf16 v8bf;
typedef __attribute__((ext_vector_type(8)))  float  v8f;

struct Params {
  const float* eeg; const float* audio;
  float* out_e; float* out_a;
  // bf16 weights, pre-shuffled to WMMA B-fragment layout (in d_ws)
  const unsigned short *sq[2], *sk[2], *sv[2], *sp[2], *f1[2], *f2[2], *fin[2];
  const unsigned short *cq_e, *ck_a, *cv_a, *lr_e, *cq_a, *ck_e, *cv_e, *lr_a;
  // fp32 biases / LN params (directly from d_in)
  const float *sq_b[2], *sk_b[2], *sv_b[2], *sp_b[2];
  const float *n1_g[2], *n1_b[2], *f1_b[2], *f2_b[2], *n2_g[2], *n2_b[2], *fin_b[2];
  const float *cq_e_b, *ck_a_b, *cv_a_b, *lr_e_b, *cq_a_b, *ck_e_b, *cv_e_b, *lr_a_b;
};

// Convert fp32 W[K][N] (row-major) -> bf16 in B-fragment order:
// dst[((kt*(N/16)+nt)*32 + lane)*16 + e], lane = ((k>>4)&1)*16 + (n&15), e = k&15.
// One contiguous 32B chunk per lane per (32K x 16N) tile -> single v16bf load in the GEMM.
__global__ void cvt_shuffle_kernel(const float* __restrict__ s, unsigned short* __restrict__ d,
                                   int K, int N) {
  int i = blockIdx.x * blockDim.x + threadIdx.x;
  if (i < K * N) {
    int k = i / N, n = i - k * N;
    union { float f; unsigned int u; } v; v.f = s[i];
    unsigned int u = v.u + 0x7FFFu + ((v.u >> 16) & 1u);   // round-to-nearest-even
    int kt = k >> 5, kk = k & 31;
    int half = kk >> 4, e = kk & 15;
    int nt = n >> 4, l16 = n & 15;
    size_t off = ((((size_t)kt * (N >> 4) + nt) << 5) + half * 16 + l16) * 16 + e;
    d[off] = (unsigned short)(u >> 16);
  }
}

// D[16 x N] (f32, LDS) = A[16 x K] (bf16, LDS) @ W (bf16, fragment-shuffled, global) + bias
// Each wave owns N/16/NWAVES tiles; one A fragment per 32-K step is shared across them.
template<int N, int K>
__device__ __forceinline__ void wave_gemm(
    const __bf16* __restrict__ A, int lda,
    const unsigned short* __restrict__ Wg,
    float* __restrict__ D, int ldd, int dcol0,
    const float* __restrict__ bias)
{
  constexpr int NT  = N >> 4;
  constexpr int TPW = NT / NWAVES;     // 1 (N=128), 2 (N=256), 4 (N=512)
  static_assert(NT % NWAVES == 0, "tile count must divide waves");
  const __bf16* W = (const __bf16*)Wg;
  const int wave = threadIdx.x >> 5;
  const int lane = threadIdx.x & 31;
  const int half = lane >> 4;
  const int l16  = lane & 15;

  v8f c[TPW] = {};
  const __bf16* Abase = A + l16 * lda + half * 16;  // 16B-aligned
#pragma unroll
  for (int kt = 0; kt < (K >> 5); ++kt) {
    // A 16x32 bf16: lanes 0-15 K=0..15, lanes 16-31 K=16..31
    v8bf alo = *(const v8bf*)(Abase + kt * 32);
    v8bf ahi = *(const v8bf*)(Abase + kt * 32 + 8);
    v16bf a = __builtin_shufflevector(alo, ahi,
        0, 1, 2, 3, 4, 5, 6, 7, 8, 9, 10, 11, 12, 13, 14, 15);
#pragma unroll
    for (int t = 0; t < TPW; ++t) {
      const int nt = wave + t * NWAVES;
      v16bf b = *(const v16bf*)(W + (((size_t)kt * NT + nt) << 9) + lane * 16); // 32B/lane
      c[t] = __builtin_amdgcn_wmma_f32_16x16x32_bf16(false, a, false, b, (short)0, c[t],
                                                     false, false);
    }
  }
#pragma unroll
  for (int t = 0; t < TPW; ++t) {
    const int tn = (wave + t * NWAVES) << 4;
    const float bn = bias[tn + l16];
    float* Dr = D + dcol0 + tn + l16;
#pragma unroll
    for (int r = 0; r < 8; ++r) Dr[(half * 8 + r) * ldd] = c[t][r] + bn; // M=r+8*half, N=l16
  }
}

// stream <- LN(stream + delta)*g + b   (optionally o = 0.3*o + 0.7*gmix), also bf16 copy -> act
__device__ __forceinline__ void ln_residual(
    float* __restrict__ stream, int lds_,
    const float* __restrict__ delta, int ldd,
    const float* __restrict__ g, const float* __restrict__ b,
    __bf16* __restrict__ act, int ldab,
    const float* __restrict__ gmix, int row0,
    float* __restrict__ s_red, float* __restrict__ s_red2,
    float* __restrict__ s_mu, float* __restrict__ s_rs)
{
  const int tid = threadIdx.x;
  const int r = tid >> 4, p = tid & 15;
  float s = 0.f, s2 = 0.f;
  for (int c = p * 16; c < p * 16 + 16; ++c) {
    float v = stream[r * lds_ + c] + delta[r * ldd + c];
    s += v; s2 += v * v;
  }
  s_red[r * 16 + p] = s; s_red2[r * 16 + p] = s2;
  __syncthreads();
  if (tid < MT) {
    float m = 0.f, q = 0.f;
    for (int i = 0; i < 16; ++i) { m += s_red[tid * 16 + i]; q += s_red2[tid * 16 + i]; }
    m *= (1.0f / ED); q = q * (1.0f / ED) - m * m;        // biased variance (jnp.var)
    s_mu[tid] = m; s_rs[tid] = rsqrtf(q + 1e-5f);
  }
  __syncthreads();
  for (int idx = tid; idx < MT * ED; idx += 256) {
    int rr = idx >> 8, c = idx & 255;
    float v = stream[rr * lds_ + c] + delta[rr * ldd + c];
    float o = (v - s_mu[rr]) * s_rs[rr] * g[c] + b[c];
    if (gmix) o = 0.3f * o + 0.7f * gmix[(size_t)(row0 + rr) * ED + c];
    stream[rr * lds_ + c] = o;
    act[rr * ldab + c] = (__bf16)o;
  }
  __syncthreads();
}

__global__ __launch_bounds__(256) void clara_fused_kernel(Params P) {
  __shared__ float  s_re[MT][LDS_S];                 // eeg residual stream (f32)
  __shared__ float  s_ra[MT][LDS_S];                 // audio residual stream (f32)
  __shared__ float  s_t [MT][LDT];                   // f32 GEMM scratch (up to 16x512)
  __shared__ __align__(16) __bf16 s_ebf[MT][LDA];    // eeg activation (bf16 A-operand)
  __shared__ __align__(16) __bf16 s_afb[MT][LDA];    // audio activation (bf16 A-operand)
  __shared__ __align__(16) __bf16 s_hbf[MT][LDH];    // bf16 staging (attn*w, cross t, FFN hidden)
  __shared__ float  s_red[MT][16], s_red2[MT][16];
  __shared__ float  s_mu[MT], s_rs[MT];
  __shared__ float  s_attw[MT][4];
  __shared__ float  s_ewe[MT], s_ewa[MT];

  const int r0  = blockIdx.x * MT;
  const int tid = threadIdx.x;

  // ---- load inputs (f32 stream + bf16 operand copies) ----
  for (int idx = tid; idx < MT * ED; idx += 256) {
    int r = idx >> 8, c = idx & 255;
    float e = P.eeg  [(size_t)(r0 + r) * ED + c];
    float a = P.audio[(size_t)(r0 + r) * ED + c];
    s_re[r][c] = e; s_ebf[r][c] = (__bf16)e;
    s_ra[r][c] = a; s_afb[r][c] = (__bf16)a;
  }
  __syncthreads();

  // ---- stage 1: self-attention (softmax over heads) + LN1, per modality ----
  for (int m = 0; m < 2; ++m) {
    float*  stream = m ? &s_ra[0][0] : &s_re[0][0];
    __bf16* act    = m ? &s_afb[0][0] : &s_ebf[0][0];
    wave_gemm<ED, ED>(act, LDA, P.sq[m], &s_t[0][0], LDT,   0, P.sq_b[m]);   // q -> [0,256)
    wave_gemm<ED, ED>(act, LDA, P.sk[m], &s_t[0][0], LDT, 256, P.sk_b[m]);   // k -> [256,512)
    __syncthreads();
    { // per-head dot(q,k): p covers 16 cols, head = p>>2
      int r = tid >> 4, p = tid & 15; float s = 0.f;
      for (int c = p * 16; c < p * 16 + 16; ++c) s += s_t[r][c] * s_t[r][256 + c];
      s_red[r][p] = s;
    }
    __syncthreads();
    if (tid < MT) {
      float w[4], mx = -1e30f;
      for (int h = 0; h < 4; ++h) {
        float d = 0.f;
        for (int p = 0; p < 4; ++p) d += s_red[tid][h * 4 + p];
        w[h] = d * 0.125f;                       // / sqrt(64)
        mx = fmaxf(mx, w[h]);
      }
      float sm = 0.f;
      for (int h = 0; h < 4; ++h) { w[h] = __expf(w[h] - mx); sm += w[h]; }
      for (int h = 0; h < 4; ++h) s_attw[tid][h] = w[h] / sm;
    }
    __syncthreads();
    wave_gemm<ED, ED>(act, LDA, P.sv[m], &s_t[0][0], LDT, 0, P.sv_b[m]);     // v -> [0,256)
    __syncthreads();
    for (int idx = tid; idx < MT * ED; idx += 256) {                          // w_h * v -> bf16
      int r = idx >> 8, c = idx & 255;
      s_hbf[r][c] = (__bf16)(s_t[r][c] * s_attw[r][c >> 6]);
    }
    __syncthreads();
    wave_gemm<ED, ED>(&s_hbf[0][0], LDH, P.sp[m], &s_t[0][0], LDT, 256, P.sp_b[m]);
    __syncthreads();
    ln_residual(stream, LDS_S, &s_t[0][256], LDT, P.n1_g[m], P.n1_b[m],
                act, LDA, nullptr, r0, &s_red[0][0], &s_red2[0][0], s_mu, s_rs);
  }

  // ---- stage 2: cross-modal low-rank interaction ----
  wave_gemm<RD, ED>(&s_ebf[0][0], LDA, P.cq_e, &s_t[0][0], LDT,   0, P.cq_e_b);  // eq
  wave_gemm<RD, ED>(&s_afb[0][0], LDA, P.ck_a, &s_t[0][0], LDT, 128, P.ck_a_b);  // ak
  wave_gemm<RD, ED>(&s_afb[0][0], LDA, P.cq_a, &s_t[0][0], LDT, 256, P.cq_a_b);  // aq
  wave_gemm<RD, ED>(&s_ebf[0][0], LDA, P.ck_e, &s_t[0][0], LDT, 384, P.ck_e_b);  // ek
  __syncthreads();
  { // t_e = eq*ak, t_a = aq*ek (bf16 for lr GEMMs) + sigmoid gate partials
    int r = tid >> 4, p = tid & 15; float se = 0.f, sa = 0.f;
    for (int c = p * 8; c < p * 8 + 8; ++c) {
      float te = s_t[r][c] * s_t[r][128 + c];
      float ta = s_t[r][256 + c] * s_t[r][384 + c];
      s_hbf[r][c] = (__bf16)te; s_hbf[r][128 + c] = (__bf16)ta;
      se += te; sa += ta;
    }
    s_red[r][p] = se; s_red2[r][p] = sa;
  }
  __syncthreads();
  if (tid < MT) {
    float se = 0.f, sa = 0.f;
    for (int p = 0; p < 16; ++p) { se += s_red[tid][p]; sa += s_red2[tid][p]; }
    s_ewe[tid] = 1.f / (1.f + __expf(-se));
    s_ewa[tid] = 1.f / (1.f + __expf(-sa));
  }
  __syncthreads();
  wave_gemm<ED, ED>(&s_afb[0][0], LDA, P.cv_a, &s_t[0][0], LDT,   0, P.cv_a_b);  // av
  wave_gemm<ED, RD>(&s_hbf[0][0], LDH, P.lr_e, &s_t[0][0], LDT, 256, P.lr_e_b);  // lr_e
  __syncthreads();
  for (int idx = tid; idx < MT * ED; idx += 256) {   // eeg_fi = eeg_as + ew*av + lr_e
    int r = idx >> 8, c = idx & 255;
    s_re[r][c] += s_ewe[r] * s_t[r][c] + s_t[r][256 + c];
  }
  __syncthreads();
  wave_gemm<ED, ED>(&s_ebf[0][0], LDA, P.cv_e, &s_t[0][0], LDT,   0, P.cv_e_b);  // ev
  wave_gemm<ED, RD>(&s_hbf[0][128], LDH, P.lr_a, &s_t[0][0], LDT, 256, P.lr_a_b);// lr_a
  __syncthreads();
  for (int idx = tid; idx < MT * ED; idx += 256) {   // aud_fi = aud_as + 0.3*cm ; refresh bf16
    int r = idx >> 8, c = idx & 255;
    s_ra[r][c] += 0.3f * (s_ewa[r] * s_t[r][c] + s_t[r][256 + c]);
    s_ebf[r][c] = (__bf16)s_re[r][c];
    s_afb[r][c] = (__bf16)s_ra[r][c];
  }
  __syncthreads();

  // ---- stage 3: FFN (exact GELU) + LN2 (+0.3/0.7 mix for audio) ----
  for (int m = 0; m < 2; ++m) {
    float*  stream = m ? &s_ra[0][0] : &s_re[0][0];
    __bf16* act    = m ? &s_afb[0][0] : &s_ebf[0][0];
    wave_gemm<FD, ED>(act, LDA, P.f1[m], &s_t[0][0], LDT, 0, P.f1_b[m]);
    __syncthreads();
    for (int idx = tid; idx < MT * FD; idx += 256) {
      int r = idx >> 9, c = idx & 511;
      float x = s_t[r][c];
      s_hbf[r][c] = (__bf16)(0.5f * x * (1.0f + erff(x * 0.70710678f)));
    }
    __syncthreads();
    wave_gemm<ED, FD>(&s_hbf[0][0], LDH, P.f2[m], &s_t[0][0], LDT, 0, P.f2_b[m]);
    __syncthreads();
    ln_residual(stream, LDS_S, &s_t[0][0], LDT, P.n2_g[m], P.n2_b[m],
                act, LDA, m ? P.audio : nullptr, r0,
                &s_red[0][0], &s_red2[0][0], s_mu, s_rs);
  }

  // ---- stage 4: final alignment projections + input residuals ----
  for (int m = 0; m < 2; ++m) {
    __bf16* act = m ? &s_afb[0][0] : &s_ebf[0][0];
    wave_gemm<ED, ED>(act, LDA, P.fin[m], &s_t[0][0], LDT, 0, P.fin_b[m]);
    __syncthreads();
    for (int idx = tid; idx < MT * ED; idx += 256) {
      int r = idx >> 8, c = idx & 255;
      size_t go = (size_t)(r0 + r) * ED + c;
      if (m == 0) P.out_e[go] = s_t[r][c] + P.eeg[go];
      else        P.out_a[go] = s_t[r][c] + 0.7f * P.audio[go];
    }
    __syncthreads();
  }
}

} // namespace

extern "C" void kernel_launch(void* const* d_in, const int* in_sizes, int n_in,
                              void* d_out, int out_size, void* d_ws, size_t ws_size,
                              hipStream_t stream) {
  (void)n_in; (void)out_size; (void)ws_size;

  // fp32 weight matrices to convert -> shuffled bf16 in d_ws (d_in index, K, N)
  const int conv_idx[22] = { 2, 4, 6, 8, 12, 14, 18,
                             20, 22, 24, 26, 30, 32, 36,
                             38, 40, 42, 44, 46, 48, 50, 52 };
  const int conv_k[22]   = { 256, 256, 256, 256, 256, 512, 256,
                             256, 256, 256, 256, 256, 512, 256,
                             256, 256, 256, 128, 256, 256, 256, 128 };
  const int conv_n[22]   = { 256, 256, 256, 256, 512, 256, 256,
                             256, 256, 256, 256, 512, 256, 256,
                             128, 128, 256, 256, 128, 128, 256, 256 };
  unsigned short* ws = (unsigned short*)d_ws;
  const unsigned short* wp[22];
  size_t off = 0;
  for (int i = 0; i < 22; ++i) {
    unsigned short* dst = ws + off;
    int n = conv_k[i] * conv_n[i];
    cvt_shuffle_kernel<<<(n + 255) / 256, 256, 0, stream>>>(
        (const float*)d_in[conv_idx[i]], dst, conv_k[i], conv_n[i]);
    wp[i] = dst;
    off += (size_t)n;
  }

  Params P;
  P.eeg   = (const float*)d_in[0];
  P.audio = (const float*)d_in[1];
  const int nrows = in_sizes[0] / 256;
  P.out_e = (float*)d_out;
  P.out_a = (float*)d_out + (size_t)nrows * 256;

  for (int m = 0; m < 2; ++m) {
    P.sq[m] = wp[m * 7 + 0]; P.sk[m] = wp[m * 7 + 1]; P.sv[m] = wp[m * 7 + 2];
    P.sp[m] = wp[m * 7 + 3]; P.f1[m] = wp[m * 7 + 4]; P.f2[m] = wp[m * 7 + 5];
    P.fin[m] = wp[m * 7 + 6];
  }
  P.cq_e = wp[14]; P.ck_a = wp[15]; P.cv_a = wp[16]; P.lr_e = wp[17];
  P.cq_a = wp[18]; P.ck_e = wp[19]; P.cv_e = wp[20]; P.lr_a = wp[21];

  for (int m = 0; m < 2; ++m) {
    int b = 2 + m * 18;  // modality base in d_in
    P.sq_b[m]  = (const float*)d_in[b + 1];
    P.sk_b[m]  = (const float*)d_in[b + 3];
    P.sv_b[m]  = (const float*)d_in[b + 5];
    P.sp_b[m]  = (const float*)d_in[b + 7];
    P.n1_g[m]  = (const float*)d_in[b + 8];
    P.n1_b[m]  = (const float*)d_in[b + 9];
    P.f1_b[m]  = (const float*)d_in[b + 11];
    P.f2_b[m]  = (const float*)d_in[b + 13];
    P.n2_g[m]  = (const float*)d_in[b + 14];
    P.n2_b[m]  = (const float*)d_in[b + 15];
    P.fin_b[m] = (const float*)d_in[b + 17];
  }
  P.cq_e_b = (const float*)d_in[39]; P.ck_a_b = (const float*)d_in[41];
  P.cv_a_b = (const float*)d_in[43]; P.lr_e_b = (const float*)d_in[45];
  P.cq_a_b = (const float*)d_in[47]; P.ck_e_b = (const float*)d_in[49];
  P.cv_e_b = (const float*)d_in[51]; P.lr_a_b = (const float*)d_in[53];

  const int nblocks = nrows / MT;
  clara_fused_kernel<<<nblocks, 256, 0, stream>>>(P);
}
